// ExpertNet_GRU_56075093016668
// MI455X (gfx1250) — matrix operand order repaired and checked by reading, not compiled, not run on hardware
//
#include <hip/hip_runtime.h>
#include <hip/hip_bf16.h>

// ---------------------------------------------------------------------------
// Problem constants (from reference)
// ---------------------------------------------------------------------------
static constexpr int B_  = 1024;
static constexpr int T_  = 128;
static constexpr int I_  = 128;
static constexpr int H_  = 256;
static constexpr int K_  = 8;
static constexpr int E1_ = 512;
static constexpr int E2_ = 256;
static constexpr int C_  = 2;
static constexpr int G3  = 3 * H_;   // 768 gates width

typedef __attribute__((ext_vector_type(16))) __bf16 bf16x16;
typedef __attribute__((ext_vector_type(8)))  __bf16 bf16x8;
typedef __attribute__((ext_vector_type(8)))  float  f32x8;
typedef __attribute__((ext_vector_type(4)))  unsigned int u32x4;
typedef __attribute__((ext_vector_type(8)))  int i32x8;
typedef __attribute__((ext_vector_type(4)))  int i32x4;

#if defined(__has_builtin)
#if __has_builtin(__builtin_amdgcn_tensor_load_to_lds) && \
    __has_builtin(__builtin_amdgcn_s_wait_tensorcnt)
#define USE_TDM 1
#endif
#endif
#ifndef USE_TDM
#define USE_TDM 0
#endif

#define DEVI __device__ __forceinline__

DEVI bf16x16 load2x8(const __bf16* p0, const __bf16* p1) {
  bf16x8 a = *reinterpret_cast<const bf16x8*>(p0);
  bf16x8 b = *reinterpret_cast<const bf16x8*>(p1);
  bf16x16 r;
#pragma unroll
  for (int i = 0; i < 8; ++i) { r[i] = a[i]; r[i + 8] = b[i]; }
  return r;
}

DEVI f32x8 wmma_bf16(bf16x16 a, bf16x16 b, f32x8 c) {
  // D = A(16x32 bf16) * B(32x16 bf16) + C(16x16 f32)
  return __builtin_amdgcn_wmma_f32_16x16x32_bf16(false, a, false, b,
                                                 (short)0, c, false, false);
}

DEVI float sigmoidf_(float x) { return 1.0f / (1.0f + __expf(-x)); }

#if USE_TDM
// TDM: copy one [16 x 768] bf16 GI slab (rows strided by B*768 bf16) into LDS.
// D# per CDNA5 ISA ch.8: group0 = {count|flags, lds_addr, gaddr_lo, gaddr_hi|type},
// group1 packs data_size/dims/strides. 2D tile, data_size = 8B units:
//   tile_dim0 = tensor_dim0 = 768*2/8 = 192, tile_dim1 = tensor_dim1 = 16,
//   tensor_dim0_stride = 1024*768*2/8 = 196608.
DEVI void tdm_load_gi_slab(const __bf16* gsrc, unsigned lds_off) {
  unsigned long long ga = (unsigned long long)(size_t)gsrc;
  u32x4 g0;
  g0[0] = 1u;                                           // count=1, user mode
  g0[1] = lds_off;                                      // LDS byte address
  g0[2] = (unsigned)(ga & 0xffffffffu);                 // global_addr[31:0]
  g0[3] = (unsigned)((ga >> 32) & 0x01ffffffu)          // global_addr[56:32]
          | 0x80000000u;                                // type=2 ("image")
  i32x8 g1;
  g1[0] = (3 << 16);                                    // data_size = 8 bytes
  g1[1] = (int)((192u & 0xffffu) << 16);                // tensor_dim0[15:0]
  g1[2] = (int)((16u & 0xffffu) << 16);                 // td0 hi | tensor_dim1 lo
  g1[3] = (int)(192u << 16);                            // td1 hi | tile_dim0
  g1[4] = 16;                                           // tile_dim1 | tile_dim2=0
  g1[5] = 196608;                                       // tensor_dim0_stride lo32
  g1[6] = 0;                                            // stride hi | dim1_stride lo
  g1[7] = 0;
  i32x4 gz4 = {0, 0, 0, 0};                             // groups 2/3 unused (2D tile)
  i32x8 gz8 = {0, 0, 0, 0, 0, 0, 0, 0};
  __builtin_amdgcn_tensor_load_to_lds(g0, g1, gz4, gz4, gz8, 0);
}
#endif

// ---------------------------------------------------------------------------
// Elementwise casts
// ---------------------------------------------------------------------------
__global__ void cast_f32_bf16_k(const float* __restrict__ src,
                                __bf16* __restrict__ dst, long n) {
  long i = (long)blockIdx.x * blockDim.x + threadIdx.x;
  if (i < n) dst[i] = (__bf16)src[i];
}

// x [B,T,I] f32  ->  xt [T,B,I] bf16
__global__ void transpose_cast_x_k(const float* __restrict__ x,
                                   __bf16* __restrict__ xt) {
  long idx = (long)blockIdx.x * blockDim.x + threadIdx.x;   // over B*T*I
  int i = (int)(idx % I_);
  long r = idx / I_;
  int t = (int)(r % T_);
  int b = (int)(r / T_);
  xt[((long)t * B_ + b) * I_ + i] = (__bf16)x[idx];
}

// ---------------------------------------------------------------------------
// Generic WMMA GEMM:  Out[m, n] = act( A[m,:Kd] . W[n,:Kd] + bias[n] )
// A row-major [M, lda] bf16, W row-major [N, Kd] bf16, Out bf16 strided.
// Batched along grid.y (expert index). One wave computes a 16x64 strip
// (4 N-tiles) so the A fragment is reused 4x per K-step.
// ---------------------------------------------------------------------------
__global__ void gemm_bf16_wmma_k(const __bf16* __restrict__ A, long long aBatch, int lda,
                                 const __bf16* __restrict__ W, long long wBatch,
                                 const float*  __restrict__ bias, int biasBatch,
                                 __bf16* __restrict__ Out, long long oBatch, int ostride,
                                 int M, int N, int Kd, int relu) {
  const int wave = threadIdx.x >> 5;
  const int lane = threadIdx.x & 31;
  const int hf   = lane >> 4;        // half-wave select
  const int l15  = lane & 15;

  const int nGrp = N >> 6;           // groups of four 16-wide tiles
  long grp = (long)blockIdx.x * 8 + wave;
  if (grp >= (long)(M >> 4) * nGrp) return;
  const int mT = (int)(grp / nGrp);
  const int nG = (int)(grp % nGrp);
  const int z  = blockIdx.y;

  const __bf16* Ab = A + (long)z * aBatch;
  const __bf16* Wb = W + (long)z * wBatch;

  const int mrow = mT * 16 + l15;    // A row this lane supplies

  f32x8 c[4];
#pragma unroll
  for (int j = 0; j < 4; ++j) c[j] = (f32x8){0, 0, 0, 0, 0, 0, 0, 0};

  for (int k0 = 0; k0 < Kd; k0 += 32) {
    const int ka = k0 + hf * 8;      // A lane K layout: {0..7,16..23} / {8..15,24..31}
    bf16x16 a = load2x8(Ab + (long)mrow * lda + ka,
                        Ab + (long)mrow * lda + ka + 16);
    const int kb = k0 + hf * 8;      // B lane K layout: {0..7,16..23} / {8..15,24..31}
#pragma unroll
    for (int j = 0; j < 4; ++j) {
      const int ncol = (nG * 4 + j) * 16 + l15;
      bf16x16 b = load2x8(Wb + (long)ncol * Kd + kb,
                          Wb + (long)ncol * Kd + kb + 16);
      c[j] = wmma_bf16(a, b, c[j]);
    }
  }

  __bf16* Ob = Out + (long)z * oBatch;
#pragma unroll
  for (int j = 0; j < 4; ++j) {
    const int ncol = (nG * 4 + j) * 16 + l15;
    const float bv = bias[(long)z * biasBatch + ncol];
#pragma unroll
    for (int v = 0; v < 8; ++v) {
      const int m = mT * 16 + v + hf * 8;
      float val = c[j][v] + bv;
      if (relu) val = fmaxf(val, 0.0f);
      Ob[(long)m * ostride + ncol] = (__bf16)val;
    }
  }
}

// ---------------------------------------------------------------------------
// GRU recurrence over T steps for a 16-row batch tile.
// GI  : [T, B, 768] bf16, already contains x@W_ih^T + b_ih
// Whh : [768, 256]  bf16 (kept resident in VGPRs across the whole t-loop)
// 16 waves; wave w owns gate columns [w*48, w*48+48) (3 WMMA N-tiles).
// GI slab for step t+1 is prefetched into double-buffered LDS by the TDM
// while step t computes; h (f32) and the gh staging buffer also live in LDS.
// ---------------------------------------------------------------------------
__global__ void __launch_bounds__(512)
gru_recurrence_k(const __bf16* __restrict__ GI,
                 const __bf16* __restrict__ Whh,
                 const float*  __restrict__ bhh,
                 __bf16* __restrict__ hOutAll,   // [T,B,H] bf16 (if storeAll)
                 float*  __restrict__ zOut,      // [B,H] f32  (if storeZ)
                 __bf16* __restrict__ zOutB,     // [B,H] bf16 (if storeZ)
                 int T, int Bn, int storeAll, int storeZ) {
  __shared__ float h_lds[16][H_ + 8];     // padded against bank conflicts
  __shared__ float gh_lds[16][G3 + 8];
#if USE_TDM
  __shared__ __bf16 gi_lds[2][16 * G3];   // double-buffered GI slab
#endif

  const int tid  = threadIdx.x;
  const int wave = tid >> 5;
  const int lane = tid & 31;
  const int hf   = lane >> 4;
  const int l15  = lane & 15;
  const int b0   = blockIdx.x * 16;

  // h = 0
  for (int idx = tid; idx < 16 * H_; idx += blockDim.x)
    h_lds[idx >> 8][idx & 255] = 0.0f;

  // Hoist W_hh tiles into registers: 3 n-tiles x 8 k-steps per wave.
  bf16x16 Bw[3][8];
#pragma unroll
  for (int j = 0; j < 3; ++j) {
    const int ncol = (wave * 3 + j) * 16 + l15;
#pragma unroll
    for (int ks = 0; ks < 8; ++ks) {
      const int kb = ks * 32 + hf * 8;
      Bw[j][ks] = load2x8(Whh + (long)ncol * H_ + kb,
                          Whh + (long)ncol * H_ + kb + 16);
    }
  }

#if USE_TDM
  // LDS byte offsets of the two GI buffers (generic ptr low 32 bits = LDS addr)
  const unsigned gi_off0 = (unsigned)(size_t)(&gi_lds[0][0]);
  const unsigned gi_off1 = (unsigned)(size_t)(&gi_lds[1][0]);
  if (wave == 0)   // prefetch slab for t = 0
    tdm_load_gi_slab(GI + ((long)0 * Bn + b0) * G3, gi_off0);
#endif
  __syncthreads();

  for (int t = 0; t < T; ++t) {
#if USE_TDM
    const int cur = t & 1;
    if (wave == 0) {
      if (t + 1 < T) {
        tdm_load_gi_slab(GI + ((long)(t + 1) * Bn + b0) * G3,
                         (t + 1) & 1 ? gi_off1 : gi_off0);
        __builtin_amdgcn_s_wait_tensorcnt(1);  // in-order: slab t complete
      } else {
        __builtin_amdgcn_s_wait_tensorcnt(0);
      }
    }
    __syncthreads();                           // gi_lds[cur] visible to all
#endif

    // ---- gh = h @ Whh^T  (pure LDS->bf16 + WMMA inner loop) ----
    f32x8 acc[3];
#pragma unroll
    for (int j = 0; j < 3; ++j) acc[j] = (f32x8){0, 0, 0, 0, 0, 0, 0, 0};

#pragma unroll
    for (int ks = 0; ks < 8; ++ks) {
      const int ka = ks * 32 + hf * 8;
      bf16x16 a;
#pragma unroll
      for (int i = 0; i < 8; ++i) {
        a[i]     = (__bf16)h_lds[l15][ka + i];
        a[i + 8] = (__bf16)h_lds[l15][ka + 16 + i];
      }
#pragma unroll
      for (int j = 0; j < 3; ++j) acc[j] = wmma_bf16(a, Bw[j][ks], acc[j]);
    }

    // ---- stage gh (+gi for r,z gates; +b_hh always) into LDS ----
#if !USE_TDM
    const __bf16* GIt = GI + ((long)t * Bn + b0) * G3;
#endif
#pragma unroll
    for (int j = 0; j < 3; ++j) {
      const int n  = (wave * 3 + j) * 16 + l15;
      const float bh = bhh[n];
#pragma unroll
      for (int v = 0; v < 8; ++v) {
        const int m = v + hf * 8;
        float val = acc[j][v] + bh;
        if (n < 2 * H_) {
#if USE_TDM
          val += (float)gi_lds[cur][m * G3 + n];             // r,z: gi+gh
#else
          val += (float)GIt[(long)m * G3 + n];
#endif
        }
        gh_lds[m][n] = val;                                   // n-gate: hn only
      }
    }
    __syncthreads();

    // ---- gates + state update ----
    for (int idx = tid; idx < 16 * H_; idx += blockDim.x) {
      const int m = idx >> 8, col = idx & 255;
      const float r  = sigmoidf_(gh_lds[m][col]);
      const float zg = sigmoidf_(gh_lds[m][col + H_]);
      const float hn = gh_lds[m][col + 2 * H_];
#if USE_TDM
      const float gin = (float)gi_lds[cur][m * G3 + 2 * H_ + col];
#else
      const float gin = (float)GIt[(long)m * G3 + 2 * H_ + col];
#endif
      const float n  = tanhf(gin + r * hn);
      const float hnew = (1.0f - zg) * n + zg * h_lds[m][col];
      h_lds[m][col] = hnew;
      if (storeAll)
        hOutAll[((long)t * Bn + b0 + m) * H_ + col] = (__bf16)hnew;
    }
    __syncthreads();
  }

  if (storeZ) {
    for (int idx = tid; idx < 16 * H_; idx += blockDim.x) {
      const int m = idx >> 8, col = idx & 255;
      const float hv = h_lds[m][col];
      zOut[(long)(b0 + m) * H_ + col]  = hv;
      zOutB[(long)(b0 + m) * H_ + col] = (__bf16)hv;
    }
  }
}

// ---------------------------------------------------------------------------
// Student-t soft assignment q[b,k] (alpha = 1)
// ---------------------------------------------------------------------------
__global__ void cluster_q_k(const float* __restrict__ z,
                            const float* __restrict__ centers,
                            float* __restrict__ q, int Bn) {
  const int b = blockIdx.x * blockDim.x + threadIdx.x;
  if (b >= Bn) return;
  float qv[K_];
  float s = 0.0f;
#pragma unroll
  for (int k = 0; k < K_; ++k) {
    float d2 = 0.0f;
    const float* zp = z + (long)b * H_;
    const float* cp = centers + (long)k * H_;
    for (int h = 0; h < H_; ++h) {
      const float d = zp[h] - cp[h];
      d2 += d * d;
    }
    qv[k] = 1.0f / (1.0f + d2);
    s += qv[k];
  }
  const float inv = 1.0f / s;
#pragma unroll
  for (int k = 0; k < K_; ++k) q[(long)b * K_ + k] = qv[k] * inv;
}

// ---------------------------------------------------------------------------
// logits = h2 @ eW3^T + eb3 ;  preds[b,c] = sum_k q[b,k] * logits[b,k,c]
// ---------------------------------------------------------------------------
__global__ void combine_k(const __bf16* __restrict__ h2,
                          const __bf16* __restrict__ eW3b,
                          const float*  __restrict__ eb3,
                          const float*  __restrict__ q,
                          float* __restrict__ preds, int Bn) {
  const int b = blockIdx.x * blockDim.x + threadIdx.x;
  if (b >= Bn) return;
  float p0 = 0.0f, p1 = 0.0f;
#pragma unroll
  for (int k = 0; k < K_; ++k) {
    float l0 = eb3[k * C_ + 0];
    float l1 = eb3[k * C_ + 1];
    const __bf16* hp = h2 + ((long)b * K_ + k) * E2_;
    const __bf16* w0 = eW3b + (long)(k * C_ + 0) * E2_;
    const __bf16* w1 = eW3b + (long)(k * C_ + 1) * E2_;
    for (int f = 0; f < E2_; ++f) {
      const float hv = (float)hp[f];
      l0 += hv * (float)w0[f];
      l1 += hv * (float)w1[f];
    }
    const float qk = q[(long)b * K_ + k];
    p0 += qk * l0;
    p1 += qk * l1;
  }
  preds[(long)b * C_ + 0] = p0;
  preds[(long)b * C_ + 1] = p1;
}

// ---------------------------------------------------------------------------
// Host launcher
// ---------------------------------------------------------------------------
extern "C" void kernel_launch(void* const* d_in, const int* in_sizes, int n_in,
                              void* d_out, int out_size, void* d_ws, size_t ws_size,
                              hipStream_t stream) {
  const float* x     = (const float*)d_in[0];
  const float* Wih0  = (const float*)d_in[1];
  const float* Whh0  = (const float*)d_in[2];
  const float* b_ih0 = (const float*)d_in[3];
  const float* b_hh0 = (const float*)d_in[4];
  const float* Wih1  = (const float*)d_in[5];
  const float* Whh1  = (const float*)d_in[6];
  const float* b_ih1 = (const float*)d_in[7];
  const float* b_hh1 = (const float*)d_in[8];
  const float* cent  = (const float*)d_in[9];
  const float* eW1   = (const float*)d_in[10];
  const float* eb1   = (const float*)d_in[11];
  const float* eW2   = (const float*)d_in[12];
  const float* eb2   = (const float*)d_in[13];
  const float* eW3   = (const float*)d_in[14];
  const float* eb3   = (const float*)d_in[15];
  float* preds = (float*)d_out;

  // ---- workspace carve-up ----
  char* w = (char*)d_ws;
  size_t off = 0;
  auto carve = [&](size_t bytes) {
    void* p = w + off;
    off = (off + bytes + 255) & ~(size_t)255;
    return p;
  };
  const long BT = (long)B_ * T_;
  __bf16* GI    = (__bf16*)carve((size_t)BT * G3 * 2);        // reused for both layers
  __bf16* xt    = (__bf16*)carve((size_t)BT * I_ * 2);
  __bf16* hl0b  = (__bf16*)carve((size_t)BT * H_ * 2);
  __bf16* Wih0b = (__bf16*)carve((size_t)G3 * I_ * 2);
  __bf16* Whh0b = (__bf16*)carve((size_t)G3 * H_ * 2);
  __bf16* Wih1b = (__bf16*)carve((size_t)G3 * H_ * 2);
  __bf16* Whh1b = (__bf16*)carve((size_t)G3 * H_ * 2);
  __bf16* eW1b  = (__bf16*)carve((size_t)K_ * E1_ * H_ * 2);
  __bf16* eW2b  = (__bf16*)carve((size_t)K_ * E2_ * E1_ * 2);
  __bf16* eW3b  = (__bf16*)carve((size_t)K_ * C_ * E2_ * 2);
  __bf16* zb    = (__bf16*)carve((size_t)B_ * H_ * 2);
  __bf16* h1b   = (__bf16*)carve((size_t)B_ * K_ * E1_ * 2);
  __bf16* h2b   = (__bf16*)carve((size_t)B_ * K_ * E2_ * 2);
  float*  zf    = (float*)carve((size_t)B_ * H_ * 4);
  float*  qv    = (float*)carve((size_t)B_ * K_ * 4);
  (void)ws_size; (void)n_in; (void)in_sizes; (void)out_size;

  auto cast = [&](const float* s, __bf16* d, long n) {
    cast_f32_bf16_k<<<dim3((unsigned)((n + 255) / 256)), 256, 0, stream>>>(s, d, n);
  };

  // 1) precision conversion
  transpose_cast_x_k<<<dim3((unsigned)(BT * I_ / 256)), 256, 0, stream>>>(x, xt);
  cast(Wih0, Wih0b, (long)G3 * I_);
  cast(Whh0, Whh0b, (long)G3 * H_);
  cast(Wih1, Wih1b, (long)G3 * H_);
  cast(Whh1, Whh1b, (long)G3 * H_);
  cast(eW1, eW1b, (long)K_ * E1_ * H_);
  cast(eW2, eW2b, (long)K_ * E2_ * E1_);
  cast(eW3, eW3b, (long)K_ * C_ * E2_);

  // 2) GI0 = xt @ W_ih0^T + b_ih0           [T*B, 768], K = 128
  {
    long tiles = (BT / 16) * (G3 / 64);
    gemm_bf16_wmma_k<<<dim3((unsigned)((tiles + 7) / 8), 1), 256, 0, stream>>>(
        xt, 0, I_, Wih0b, 0, b_ih0, 0, GI, 0, G3, (int)BT, G3, I_, 0);
  }
  // 3) layer-0 recurrence -> h_l0 (bf16, all steps)
  gru_recurrence_k<<<dim3(B_ / 16), 512, 0, stream>>>(
      GI, Whh0b, b_hh0, hl0b, nullptr, nullptr, T_, B_, 1, 0);

  // 4) GI1 = h_l0 @ W_ih1^T + b_ih1 (reuses GI buffer)   K = 256
  {
    long tiles = (BT / 16) * (G3 / 64);
    gemm_bf16_wmma_k<<<dim3((unsigned)((tiles + 7) / 8), 1), 256, 0, stream>>>(
        hl0b, 0, H_, Wih1b, 0, b_ih1, 0, GI, 0, G3, (int)BT, G3, H_, 0);
  }
  // 5) layer-1 recurrence -> z (last step only)
  gru_recurrence_k<<<dim3(B_ / 16), 512, 0, stream>>>(
      GI, Whh1b, b_hh1, nullptr, zf, zb, T_, B_, 0, 1);

  // 6) soft cluster assignment
  cluster_q_k<<<dim3(B_ / 256), 256, 0, stream>>>(zf, cent, qv, B_);

  // 7) experts: h1 = relu(z @ eW1[k]^T + eb1[k])   [B, K, E1]
  {
    long tiles = (B_ / 16) * (E1_ / 64);
    gemm_bf16_wmma_k<<<dim3((unsigned)((tiles + 7) / 8), K_), 256, 0, stream>>>(
        zb, 0, H_, eW1b, (long long)E1_ * H_, eb1, E1_,
        h1b, E1_, K_ * E1_, B_, E1_, H_, 1);
  }
  // 8) h2 = relu(h1 @ eW2[k]^T + eb2[k])           [B, K, E2]
  {
    long tiles = (B_ / 16) * (E2_ / 64);
    gemm_bf16_wmma_k<<<dim3((unsigned)((tiles + 7) / 8), K_), 256, 0, stream>>>(
        h1b, E1_, K_ * E1_, eW2b, (long long)E2_ * E1_, eb2, E2_,
        h2b, E2_, K_ * E2_, B_, E2_, E1_, 1);
  }
  // 9) logits + q-weighted combine
  combine_k<<<dim3(B_ / 256), 256, 0, stream>>>(h2b, eW3b, eb3, qv, preds, B_);
}
